// sLSTMCell_87325275062884
// MI455X (gfx1250) — matrix operand order repaired and checked by reading, not compiled, not run on hardware
//
#include <hip/hip_runtime.h>

typedef __attribute__((ext_vector_type(16))) __bf16    v16bf;
typedef __attribute__((ext_vector_type(8)))  float     v8f;
typedef __attribute__((ext_vector_type(8)))  unsigned  v8u;

constexpr int S_LEN = 1024;
constexpr int HEADS = 8;
constexpr int DDIM  = 128;
constexpr int GATES = 4;
constexpr int NCOLS = 512;           // G*D
constexpr int MROWS = 16;            // batch rows per workgroup
constexpr int GE    = 4096;          // G*E (row length of x)
constexpr size_t OUT_SEQ_ELEMS = 32ull * HEADS * S_LEN * DDIM; // 33554432

__global__ void n_allzero_kernel(const float* __restrict__ n, int* __restrict__ flag) {
  int idx = blockIdx.x * blockDim.x + threadIdx.x;   // 0..16383
  int nz = 0;
  for (int i = idx; i < 32768; i += 16384) nz |= (n[i] != 0.0f);
  if (nz) atomicOr(flag, 1);
}

// CDNA5 async copy: global -> LDS, 16 bytes per lane, tracked by ASYNCcnt.
__device__ __forceinline__ void async_load_b128(unsigned lds_byte_addr,
                                                unsigned voff_bytes,
                                                const float* sbase) {
  asm volatile("global_load_async_to_lds_b128 %0, %1, %2"
               :: "v"(lds_byte_addr), "v"(voff_bytes), "s"(sbase)
               : "memory");
}
__device__ __forceinline__ void wait_async0() {
  asm volatile("s_wait_asynccnt 0" ::: "memory");
}

__global__ void __launch_bounds__(256, 1)
slstm_scan_kernel(const float* __restrict__ x,
                  const float* __restrict__ state,
                  const float* __restrict__ R,
                  const float* __restrict__ bias,
                  float* __restrict__ out,
                  const int* __restrict__ nz_flag)
{
  __shared__ __align__(32) __bf16 aFrag[4][32][16];        // 4 KB: h_{t-1} as bf16 A fragments
  __shared__ __align__(16) float  rawbuf[MROWS][NCOLS];    // 32 KB: Ry accumulator
  extern __shared__ float xbuf[];                          // 64 KB: [2][16][512] double-buffered x

  const int head   = blockIdx.x >> 1;
  const int bhalf  = blockIdx.x & 1;
  const int tid    = threadIdx.x;
  const int wave   = tid >> 5;
  const int lane   = tid & 31;
  const int laneN  = lane & 15;
  const int laneHi = lane >> 4;

  // ---------------- Load R[head] once, pre-swizzled bf16 B fragments (VGPR-resident) ----------
  // B 32x16 bf16 layout: lanes 0-15 hold K=0..7,16..23; lanes 16-31 hold K=8..15,24..31.
  v16bf bfrag[4][4];                         // [n-tile jj][k-chunk kc]
  {
    const float* Rh = R + (size_t)head * DDIM * NCOLS;
    #pragma unroll
    for (int jj = 0; jj < 4; ++jj) {
      const int n = (wave * 4 + jj) * 16 + laneN;
      #pragma unroll
      for (int kc = 0; kc < 4; ++kc) {
        v16bf f;
        #pragma unroll
        for (int e = 0; e < 16; ++e) {
          const int k = kc * 32 + laneHi * 8 + e + (e & 8);
          f[e] = (__bf16)Rh[k * NCOLS + n];
        }
        // opaque barrier: force bf16 materialization NOW, kill the f32 copies
        v8u raw = __builtin_bit_cast(v8u, f);
        asm volatile("" : "+v"(raw));
        bfrag[jj][kc] = __builtin_bit_cast(v16bf, raw);
      }
    }
  }

  // ---------------- Gate-phase ownership: thread -> (batch row, 8 consecutive d) --------------
  const int b_local = tid >> 4;              // 0..15
  const int dbase   = (tid & 15) * 8;        // 0,8,...,120
  const int b_glob  = bhalf * 16 + b_local;

  float c_s[8], n_s[8], m_s[8], h_s[8];
  {
    const int soff = b_glob * 1024 + head * DDIM + dbase;
    #pragma unroll
    for (int q = 0; q < 8; ++q) {
      h_s[q] = state[soff + q];
      c_s[q] = state[32768 + soff + q];
      n_s[q] = state[65536 + soff + q];
      m_s[q] = state[98304 + soff + q];
    }
  }

  // seed A fragment with h_0
  #pragma unroll
  for (int q = 0; q < 8; ++q) {
    const int d = dbase + q;
    const int kc = d >> 5, kd = d & 31;
    const int group = kd >> 3;
    const int kh = group & 1;
    const int v = ((group >> 1) << 2) + ((kd & 7) >> 1);
    const int e = 2 * v + (kd & 1);
    aFrag[kc][kh * 16 + b_local][e] = (__bf16)h_s[q];
  }

  float bias_r[GATES][8];
  {
    const float* bb = bias + head * (GATES * DDIM) + dbase;
    #pragma unroll
    for (int g = 0; g < GATES; ++g)
      #pragma unroll
      for (int q = 0; q < 8; ++q) bias_r[g][q] = bb[g * DDIM + q];
  }

  // async x staging: this thread's slots in the double buffer
  const unsigned voff_base = (unsigned)(b_glob * S_LEN * GE + head * DDIM + dbase); // elements
  float* xrow0 = &xbuf[(0 * MROWS + b_local) * NCOLS + dbase];
  float* xrow1 = &xbuf[(1 * MROWS + b_local) * NCOLS + dbase];

  // issue async loads for t = 0 into buffer 0
  {
    const float* xstep = x;                  // + 0*GE
    #pragma unroll
    for (int g = 0; g < GATES; ++g)
      #pragma unroll
      for (int c = 0; c < 2; ++c)
        async_load_b128((unsigned)(uintptr_t)(xrow0 + g * DDIM + c * 4),
                        (voff_base + (unsigned)(g * 1024 + c * 4)) * 4u, xstep);
  }

  float* outp = out + ((size_t)b_glob * HEADS + head) * S_LEN * DDIM + dbase;
  const float* xpre = x + voff_base;         // for L2 prefetch 2 steps ahead
  const bool allz0 = nz_flag ? (nz_flag[0] == 0) : true;

  for (int t = 0; t < S_LEN; ++t) {
    __syncthreads();                               // aFrag ready (init or previous gate phase)

    // ------------------ matmul phase: Ry = h_{t-1} @ R[head] via bf16 WMMA ------------------
    v16bf afr[4];
    #pragma unroll
    for (int kc = 0; kc < 4; ++kc)
      afr[kc] = *(const v16bf*)&aFrag[kc][lane][0];

    #pragma unroll
    for (int jj = 0; jj < 4; ++jj) {
      v8f acc = {};
      #pragma unroll
      for (int kc = 0; kc < 4; ++kc)
        acc = __builtin_amdgcn_wmma_f32_16x16x32_bf16(
                  false, afr[kc], false, bfrag[jj][kc], (short)0, acc, false, false);
      // scatter C tile: M = v + 8*(lane/16), N = lane%16 (+ tile offset)
      const int n = (wave * 4 + jj) * 16 + laneN;
      const int mbase = laneHi * 8;
      #pragma unroll
      for (int v = 0; v < 8; ++v)
        rawbuf[mbase + v][n] = acc[v];
    }
    __syncthreads();                               // rawbuf ready; aFrag free to overwrite

    // ------------------ gate phase ------------------
    // L2 prefetch 2 steps ahead of the async engine
    if (t + 2 < S_LEN) {
      const float* xn = xpre + (size_t)(t + 2) * GE;
      #pragma unroll
      for (int g = 0; g < GATES; ++g)
        __builtin_prefetch(xn + g * 1024, 0, 1);
    }

    wait_async0();                                 // this thread's x_t slots are in LDS
    const float* xcur = (t & 1) ? xrow1 : xrow0;

    float rawv[GATES][8];
    #pragma unroll
    for (int g = 0; g < GATES; ++g) {
      const float4 r0 = *(const float4*)&rawbuf[b_local][g * DDIM + dbase];
      const float4 r1 = *(const float4*)&rawbuf[b_local][g * DDIM + dbase + 4];
      const float4 x0 = *(const float4*)&xcur[g * DDIM + 0];
      const float4 x1 = *(const float4*)&xcur[g * DDIM + 4];
      rawv[g][0] = r0.x + x0.x; rawv[g][1] = r0.y + x0.y;
      rawv[g][2] = r0.z + x0.z; rawv[g][3] = r0.w + x0.w;
      rawv[g][4] = r1.x + x1.x; rawv[g][5] = r1.y + x1.y;
      rawv[g][6] = r1.z + x1.z; rawv[g][7] = r1.w + x1.w;
    }

    // issue async loads for t+1 into the other buffer
    if (t + 1 < S_LEN) {
      const float* xstep = x + (size_t)(t + 1) * GE;
      float* xnext = (t & 1) ? xrow0 : xrow1;
      #pragma unroll
      for (int g = 0; g < GATES; ++g)
        #pragma unroll
        for (int c = 0; c < 2; ++c)
          async_load_b128((unsigned)(uintptr_t)(xnext + g * DDIM + c * 4),
                          (voff_base + (unsigned)(g * 1024 + c * 4)) * 4u, xstep);
    }

    const bool use_iraw = (t == 0) && allz0;
    #pragma unroll
    for (int q = 0; q < 8; ++q) {
      const float iraw = rawv[0][q] + bias_r[0][q];
      const float fraw = rawv[1][q] + bias_r[1][q];
      const float zraw = rawv[2][q] + bias_r[2][q];
      const float oraw = rawv[3][q] + bias_r[3][q];
      // stable log_sigmoid
      const float ls   = fminf(fraw, 0.0f) - __logf(1.0f + __expf(-fabsf(fraw)));
      const float lfm  = m_s[q] + ls;
      const float mnew = use_iraw ? iraw : fmaxf(iraw, lfm);
      const float og   = 1.0f / (1.0f + __expf(-oraw));
      const float ig   = fminf(__expf(iraw - mnew), 1.0f);
      const float fg   = fminf(__expf(lfm - mnew), 1.0f);
      const float zg   = tanhf(zraw);
      const float cn   = fg * c_s[q] + ig * zg;
      const float nn   = fg * n_s[q] + ig;
      h_s[q] = og * cn / nn;
      c_s[q] = cn; n_s[q] = nn; m_s[q] = mnew;
    }

    // write h_t to output (B,H,S,D)
    float4 o0 = { h_s[0], h_s[1], h_s[2], h_s[3] };
    float4 o1 = { h_s[4], h_s[5], h_s[6], h_s[7] };
    *(float4*)&outp[(size_t)t * DDIM]     = o0;
    *(float4*)&outp[(size_t)t * DDIM + 4] = o1;

    // publish h_t as next A fragment
    #pragma unroll
    for (int q = 0; q < 8; ++q) {
      const int d = dbase + q;
      const int kc = d >> 5, kd = d & 31;
      const int group = kd >> 3;
      const int kh = group & 1;
      const int v = ((group >> 1) << 2) + ((kd & 7) >> 1);
      const int e = 2 * v + (kd & 1);
      aFrag[kc][kh * 16 + b_local][e] = (__bf16)h_s[q];
    }
  }

  // ---------------- final state (4, B, E) ----------------
  float* fs = out + OUT_SEQ_ELEMS;
  const int soff = b_glob * 1024 + head * DDIM + dbase;
  #pragma unroll
  for (int q = 0; q < 8; ++q) {
    fs[soff + q]          = h_s[q];
    fs[32768 + soff + q]  = c_s[q];
    fs[65536 + soff + q]  = n_s[q];
    fs[98304 + soff + q]  = m_s[q];
  }
}

extern "C" void kernel_launch(void* const* d_in, const int* in_sizes, int n_in,
                              void* d_out, int out_size, void* d_ws, size_t ws_size,
                              hipStream_t stream) {
  const float* x     = (const float*)d_in[0];
  const float* state = (const float*)d_in[1];
  const float* R     = (const float*)d_in[2];
  const float* bias  = (const float*)d_in[3];
  float* out = (float*)d_out;

  int* flag = nullptr;
  if (d_ws && ws_size >= sizeof(int)) {
    flag = (int*)d_ws;
    hipMemsetAsync(flag, 0, sizeof(int), stream);
    n_allzero_kernel<<<64, 256, 0, stream>>>(state + 65536, flag);  // state[2] == n
  }

  const size_t dyn_lds = 2ull * MROWS * NCOLS * sizeof(float);      // 64 KB x double-buffer
  slstm_scan_kernel<<<16, 256, dyn_lds, stream>>>(x, state, R, bias, out, flag);
}
